// CausalSelfAttention_1941325218211
// MI455X (gfx1250) — matrix operand order repaired and checked by reading, not compiled, not run on hardware
//
#include <hip/hip_runtime.h>
#include <hip/hip_bf16.h>

// CDNA5 / gfx1250 causal self-attention, bf16 WMMA pipeline, v3.
//   prep:  x -> bf16, w_attn -> bf16 transposed [3C][C], w_proj -> bf16 [C][C]T
//   qkv :  ASYNC-to-LDS staged weight tiles shared by 8 waves; writes q,k
//          [B,H,T,64] and V transposed [B,H,64,T] (all bf16)
//   attn:  flash attention; 8 query-tiles/block share ASYNC-staged K / V^T
//   proj:  ASYNC-to-LDS staged weight tiles, fp32 output + bias

typedef __attribute__((ext_vector_type(16))) __bf16 v16bf;
typedef __attribute__((ext_vector_type(8)))  __bf16 v8bf;
typedef __attribute__((ext_vector_type(8)))  float  v8f;

#define BB 4
#define TT 2048
#define CC 1024
#define HH 16
#define HD 64
#define N3 (3*CC)
#define MM (BB*TT)

__device__ __forceinline__ v8f wmma_bf16(v16bf a, v16bf b, v8f c) {
  return __builtin_amdgcn_wmma_f32_16x16x32_bf16(false, a, false, b,
                                                 (short)0, c, false, false);
}

// Wave-relative LDS byte offset (generic LDS pointer truncates to addr[31:0]).
__device__ __forceinline__ unsigned lds_off(const void* p) {
  return (unsigned)(unsigned long long)p;
}

// Direct memory -> LDS async copy (16B), tracked with ASYNCcnt (no VGPR hop).
__device__ __forceinline__ void async_copy_b128(unsigned dst_lds, const void* src) {
  asm volatile("global_load_async_to_lds_b128 %0, %1, off"
               :: "v"(dst_lds), "v"(src) : "memory");
}
__device__ __forceinline__ void wait_async0() {
  asm volatile("s_wait_asynccnt 0" ::: "memory");
}

// Load a 16x32 A fragment: row = m (lane 0..15), k = 8*lhalf + e%8 + 16*(e/8).
__device__ __forceinline__ v16bf load_a_frag(const __bf16* row8) {
  v8bf lo = *(const v8bf*)(row8);
  v8bf hi = *(const v8bf*)(row8 + 16);
  v16bf a;
#pragma unroll
  for (int e = 0; e < 8; ++e) { a[e] = lo[e]; a[8 + e] = hi[e]; }
  return a;
}

// Build a B fragment from two 8-element bf16 runs at p and p+16 (k-major source).
__device__ __forceinline__ v16bf load_b_frag16(const __bf16* p) {
  v8bf lo = *(const v8bf*)(p);
  v8bf hi = *(const v8bf*)(p + 16);
  v16bf b;
#pragma unroll
  for (int e = 0; e < 8; ++e) { b[e] = lo[e]; b[8 + e] = hi[e]; }
  return b;
}

// ---------------------------------------------------------------------------
// Prep kernels (one-time conversions; bandwidth-trivial vs the GEMMs).
// ---------------------------------------------------------------------------
__global__ __launch_bounds__(256) void cvt_f32_bf16_kernel(
    const float* __restrict__ s, __bf16* __restrict__ d, int n)
{
  const int i = (blockIdx.x * 256 + threadIdx.x) * 4;
  if (i + 3 < n) {
    const float4 f = *(const float4*)(s + i);
    d[i + 0] = (__bf16)f.x; d[i + 1] = (__bf16)f.y;
    d[i + 2] = (__bf16)f.z; d[i + 3] = (__bf16)f.w;
  }
}

// wT[n][k] = (bf16) w[k][n]   (w is [K][N] row-major)
__global__ __launch_bounds__(256) void transpose_cvt_kernel(
    const float* __restrict__ w, __bf16* __restrict__ wT, int K, int N)
{
  const int g  = blockIdx.x * 256 + threadIdx.x;
  const int n  = g % N;
  const int k0 = (g / N) * 8;
  if (k0 >= K) return;
  const float* src = w + (size_t)k0 * N + n;   // coalesced reads across threads
  v8bf o;
#pragma unroll
  for (int j = 0; j < 8; ++j) o[j] = (__bf16)src[(size_t)j * N];
  *(v8bf*)(wT + (size_t)n * K + k0) = o;
}

// ---------------------------------------------------------------------------
// Kernel: qkv = xb @ w_attn + b_attn. Block = 256 rows x 64 cols; wave = 32x64.
// Weight tile (64 cols x 32 k, bf16) async-staged in LDS, shared by 8 waves.
// ---------------------------------------------------------------------------
__global__ __launch_bounds__(256) void qkv_gemm_kernel(
    const __bf16* __restrict__ xb, const __bf16* __restrict__ wT,
    const float* __restrict__ bias,
    __bf16* __restrict__ dq, __bf16* __restrict__ dk, __bf16* __restrict__ dvT)
{
  __shared__ __bf16 lw[64][40];                 // 64 cols x 32 k (+pad, 16B ok)

  const int lane  = threadIdx.x & 31;
  const int lmod  = lane & 15;
  const int lhalf = lane >> 4;
  const int wl    = threadIdx.x >> 5;
  const int ns    = blockIdx.x % (N3 / 64);     // 0..47 column strip
  const int rb    = blockIdx.x / (N3 / 64);     // 0..31 row block
  const int m0    = rb * 256 + wl * 32;
  const int n0    = ns * 64;

  const int nr = threadIdx.x >> 2;              // staging: 64 rows x 4 x 16B
  const int kc = (threadIdx.x & 3) * 8;
  const unsigned stage_dst = lds_off(&lw[nr][kc]);
  const __bf16* stage_src0 = wT + (size_t)(n0 + nr) * CC + kc;

  v8f acc[2][4] = {};

  for (int kb = 0; kb < CC; kb += 32) {
    const __bf16* src = stage_src0 + kb;
    async_copy_b128(stage_dst, src);            // memory -> LDS, ASYNCcnt
    __builtin_prefetch(src + 32, 0, 1);         // next K tile
    wait_async0();
    __syncthreads();

    v16bf a0 = load_a_frag(xb + (size_t)(m0 + lmod) * CC + kb + 8 * lhalf);
    v16bf a1 = load_a_frag(xb + (size_t)(m0 + 16 + lmod) * CC + kb + 8 * lhalf);
#pragma unroll
    for (int nt = 0; nt < 4; ++nt) {
      v16bf bf = load_b_frag16(&lw[nt * 16 + lmod][8 * lhalf]);
      acc[0][nt] = wmma_bf16(a0, bf, acc[0][nt]);
      acc[1][nt] = wmma_bf16(a1, bf, acc[1][nt]);
    }
    __syncthreads();
  }

  const int b  = m0 / TT;
  const int t0 = m0 % TT;
#pragma unroll
  for (int nt = 0; nt < 4; ++nt) {
    const int n   = n0 + nt * 16 + lmod;
    const int sel = n >> 10;                    // 0=q 1=k 2=v (uniform per tile)
    const int cc  = n & (CC - 1);
    const int h   = cc >> 6;
    const int col = cc & (HD - 1);
    const float bv = bias[n];
#pragma unroll
    for (int rr = 0; rr < 2; ++rr) {
      const int tb = t0 + rr * 16 + 8 * lhalf;
      if (sel == 2) {                           // V stored transposed [hd][T]
        const size_t base = ((size_t)(b * HH + h) * HD + col) * TT;
        v8bf ov;
#pragma unroll
        for (int r = 0; r < 8; ++r) ov[r] = (__bf16)(acc[rr][nt][r] + bv);
        *(v8bf*)(dvT + base + tb) = ov;
      } else {
        __bf16* dst = (sel == 0) ? dq : dk;     // [T][hd] per (b,h)
        const size_t base = (size_t)(b * HH + h) * TT * HD + col;
#pragma unroll
        for (int r = 0; r < 8; ++r)
          dst[base + (size_t)(tb + r) * HD] = (__bf16)(acc[rr][nt][r] + bv);
      }
    }
  }
}

// ---------------------------------------------------------------------------
// Kernel: flash attention. Block = one (b,h) x 8 query tiles (128 queries);
// waves share ASYNC-staged K (32x64) and V^T (64x32) tiles per key block.
// Uniform trip count over the block; causal mask kills out-of-range blocks.
// ---------------------------------------------------------------------------
__global__ __launch_bounds__(256) void attn_kernel(
    const __bf16* __restrict__ q, const __bf16* __restrict__ k,
    const __bf16* __restrict__ vT, __bf16* __restrict__ y)
{
  __shared__ __bf16 lds_k[32][72];              // [key][hd] (+pad, 16B aligned)
  __shared__ __bf16 lds_vt[64][40];             // [hd][key] (+pad)
  __shared__ __bf16 lp[8][16][34];              // per-wave 16x32 P tile

  const int lane  = threadIdx.x & 31;
  const int lmod  = lane & 15;
  const int lhalf = lane >> 4;
  const int wl    = threadIdx.x >> 5;
  const int bh    = blockIdx.x >> 4;            // 0..63
  const int qt0   = (blockIdx.x & 15) * 8;
  const int q0    = (qt0 + wl) * 16;

  const __bf16* qp  = q  + (size_t)bh * TT * HD;
  const __bf16* kp  = k  + (size_t)bh * TT * HD;
  const __bf16* vtp = vT + (size_t)bh * HD * TT;

  // Q A-fragments (hd = 2 x K32), held in registers for the whole key loop.
  const __bf16* qrow = qp + (size_t)(q0 + lmod) * HD + 8 * lhalf;
  const v16bf aq0 = load_a_frag(qrow);
  const v16bf aq1 = load_a_frag(qrow + 32);

  // Staging assignments (per thread: one 16B chunk of K, one of V^T).
  const int kr = threadIdx.x >> 3;              // 0..31 key row
  const int kcc = (threadIdx.x & 7) * 8;        // 0..56 hd col
  const unsigned kdst = lds_off(&lds_k[kr][kcc]);
  const __bf16* ksrc0 = kp + (size_t)kr * HD + kcc;
  const int vr = threadIdx.x >> 2;              // 0..63 hd row
  const int vc = (threadIdx.x & 3) * 8;         // 0..24 key col
  const unsigned vdst = lds_off(&lds_vt[vr][vc]);
  const __bf16* vsrc0 = vtp + (size_t)vr * TT + vc;

  float mrow[8], lrow[8];
  v8f o[4] = {};
#pragma unroll
  for (int r = 0; r < 8; ++r) { mrow[r] = -3.0e38f; lrow[r] = 0.0f; }

  const float scale = 0.125f;                   // 1/sqrt(64)
  const int nkb = ((qt0 + 7) * 16 + 15) / 32 + 1;   // block-uniform

  for (int jb = 0; jb < nkb; ++jb) {
    const int kb0 = jb * 32;

    // cooperative async stage: K tile 32x64, V^T tile 64x32 (both bf16)
    const __bf16* ks = ksrc0 + (size_t)kb0 * HD;
    const __bf16* vs = vsrc0 + kb0;
    async_copy_b128(kdst, ks);
    async_copy_b128(vdst, vs);
    __builtin_prefetch(ks + 32 * HD, 0, 0);
    __builtin_prefetch(vs + 32, 0, 0);
    wait_async0();
    __syncthreads();

    // ---- S = Q K^T : 16 queries x 32 keys (two 16x16 tiles) ----
    v8f s0 = {}, s1 = {};
    s0 = wmma_bf16(aq0, load_b_frag16(&lds_k[lmod][8 * lhalf]),      s0);
    s0 = wmma_bf16(aq1, load_b_frag16(&lds_k[lmod][32 + 8 * lhalf]), s0);
    s1 = wmma_bf16(aq0, load_b_frag16(&lds_k[16 + lmod][8 * lhalf]),      s1);
    s1 = wmma_bf16(aq1, load_b_frag16(&lds_k[16 + lmod][32 + 8 * lhalf]), s1);

    // ---- causal mask + scale ----
    float p0a[8], p1a[8];
#pragma unroll
    for (int r = 0; r < 8; ++r) {
      const int qi = q0 + r + 8 * lhalf;
      p0a[r] = (kb0 + lmod      <= qi) ? s0[r] * scale : -3.0e38f;
      p1a[r] = (kb0 + 16 + lmod <= qi) ? s1[r] * scale : -3.0e38f;
    }

    // ---- online softmax (row = VGPR idx; 16 columns across a lane group) ----
#pragma unroll
    for (int r = 0; r < 8; ++r) {
      float m = fmaxf(p0a[r], p1a[r]);
      m = fmaxf(m, __shfl_xor(m, 1, 32));
      m = fmaxf(m, __shfl_xor(m, 2, 32));
      m = fmaxf(m, __shfl_xor(m, 4, 32));
      m = fmaxf(m, __shfl_xor(m, 8, 32));
      const float mnew = fmaxf(mrow[r], m);
      const float corr = __expf(mrow[r] - mnew);
      mrow[r] = mnew;
      const float e0 = __expf(p0a[r] - mnew);
      const float e1 = __expf(p1a[r] - mnew);
      p0a[r] = e0; p1a[r] = e1;
      float s = e0 + e1;
      s += __shfl_xor(s, 1, 32);
      s += __shfl_xor(s, 2, 32);
      s += __shfl_xor(s, 4, 32);
      s += __shfl_xor(s, 8, 32);
      lrow[r] = lrow[r] * corr + s;
      o[0][r] *= corr; o[1][r] *= corr; o[2][r] *= corr; o[3][r] *= corr;
    }

    // ---- P: C-layout -> per-wave LDS (16x32 row-major) -> A-layout ----
#pragma unroll
    for (int r = 0; r < 8; ++r) {
      lp[wl][r + 8 * lhalf][lmod]      = (__bf16)p0a[r];
      lp[wl][r + 8 * lhalf][16 + lmod] = (__bf16)p1a[r];
    }
    __builtin_amdgcn_wave_barrier();
    asm volatile("s_wait_dscnt 0" ::: "memory");
    __builtin_amdgcn_wave_barrier();

    v16bf ap;
    {
      const __bf16* pr = &lp[wl][lmod][0] + 8 * lhalf;
#pragma unroll
      for (int e = 0; e < 8; ++e) { ap[e] = pr[e]; ap[8 + e] = pr[16 + e]; }
    }

    // ---- O += P V (K = 32 keys, four 16-col tiles from staged V^T) ----
#pragma unroll
    for (int c = 0; c < 4; ++c)
      o[c] = wmma_bf16(ap, load_b_frag16(&lds_vt[c * 16 + lmod][8 * lhalf]), o[c]);

    __syncthreads();
  }

  // ---- epilogue: normalize, write y[b][t][h*64+col] bf16 ----
  const int b = bh / HH, h = bh % HH;
#pragma unroll
  for (int r = 0; r < 8; ++r) {
    const float inv = 1.0f / lrow[r];
    const int t = q0 + r + 8 * lhalf;
    __bf16* yr = y + ((size_t)b * TT + t) * CC + h * HD;
#pragma unroll
    for (int c = 0; c < 4; ++c) yr[c * 16 + lmod] = (__bf16)(o[c][r] * inv);
  }
}

// ---------------------------------------------------------------------------
// Kernel: out = y @ w_proj + b_proj (bf16 inputs, fp32 out). Same LDS scheme.
// ---------------------------------------------------------------------------
__global__ __launch_bounds__(256) void proj_gemm_kernel(
    const __bf16* __restrict__ yb, const __bf16* __restrict__ wT,
    const float* __restrict__ bias, float* __restrict__ out)
{
  __shared__ __bf16 lw[64][40];

  const int lane  = threadIdx.x & 31;
  const int lmod  = lane & 15;
  const int lhalf = lane >> 4;
  const int wl    = threadIdx.x >> 5;
  const int ns    = blockIdx.x % (CC / 64);     // 0..15
  const int rb    = blockIdx.x / (CC / 64);     // 0..31
  const int m0    = rb * 256 + wl * 32;
  const int n0    = ns * 64;

  const int nr = threadIdx.x >> 2;
  const int kc = (threadIdx.x & 3) * 8;
  const unsigned stage_dst = lds_off(&lw[nr][kc]);
  const __bf16* stage_src0 = wT + (size_t)(n0 + nr) * CC + kc;

  v8f acc[2][4] = {};

  for (int kb = 0; kb < CC; kb += 32) {
    const __bf16* src = stage_src0 + kb;
    async_copy_b128(stage_dst, src);
    __builtin_prefetch(src + 32, 0, 1);
    wait_async0();
    __syncthreads();

    v16bf a0 = load_a_frag(yb + (size_t)(m0 + lmod) * CC + kb + 8 * lhalf);
    v16bf a1 = load_a_frag(yb + (size_t)(m0 + 16 + lmod) * CC + kb + 8 * lhalf);
#pragma unroll
    for (int nt = 0; nt < 4; ++nt) {
      v16bf bf = load_b_frag16(&lw[nt * 16 + lmod][8 * lhalf]);
      acc[0][nt] = wmma_bf16(a0, bf, acc[0][nt]);
      acc[1][nt] = wmma_bf16(a1, bf, acc[1][nt]);
    }
    __syncthreads();
  }

#pragma unroll
  for (int nt = 0; nt < 4; ++nt) {
    const int n = n0 + nt * 16 + lmod;
    const float bv = bias[n];
#pragma unroll
    for (int rr = 0; rr < 2; ++rr)
#pragma unroll
      for (int r = 0; r < 8; ++r)
        out[(size_t)(m0 + rr * 16 + r + 8 * lhalf) * CC + n] = acc[rr][nt][r] + bv;
  }
}

// ---------------------------------------------------------------------------
extern "C" void kernel_launch(void* const* d_in, const int* in_sizes, int n_in,
                              void* d_out, int out_size, void* d_ws, size_t ws_size,
                              hipStream_t stream) {
  (void)in_sizes; (void)n_in; (void)out_size; (void)ws_size;
  const float* x      = (const float*)d_in[0];
  const float* w_attn = (const float*)d_in[1];
  const float* b_attn = (const float*)d_in[2];
  const float* w_proj = (const float*)d_in[3];
  const float* b_proj = (const float*)d_in[4];
  float* out = (float*)d_out;

  // Workspace layout (bf16): xb[M*C] | wTa[3C*C] | wTp[C*C] | q | k | vT | y
  const size_t XN = (size_t)MM * CC;             // 8388608
  const size_t HQ = (size_t)BB * HH * TT * HD;   // 8388608
  __bf16* xb  = (__bf16*)d_ws;
  __bf16* wTa = xb  + XN;
  __bf16* wTp = wTa + (size_t)N3 * CC;
  __bf16* dq  = wTp + (size_t)CC * CC;
  __bf16* dk  = dq + HQ;
  __bf16* dvT = dk + HQ;
  __bf16* dy  = dvT + HQ;

  // --- prep: one-time conversions/transposes ---
  cvt_f32_bf16_kernel<<<(int)(XN / 4 / 256), 256, 0, stream>>>(x, xb, (int)XN);
  transpose_cvt_kernel<<<(CC / 8) * N3 / 256, 256, 0, stream>>>(w_attn, wTa, CC, N3);
  transpose_cvt_kernel<<<(CC / 8) * CC / 256, 256, 0, stream>>>(w_proj, wTp, CC, CC);

  // --- qkv GEMM: 32 row-blocks x 48 column strips ---
  qkv_gemm_kernel<<<(MM / 256) * (N3 / 64), 256, 0, stream>>>(
      xb, wTa, b_attn, dq, dk, dvT);

  // --- flash attention: (B*H)=64 x 16 query-tile groups ---
  attn_kernel<<<(BB * HH) * (TT / 128), 256, 0, stream>>>(dq, dk, dvT, dy);

  // --- output projection: 32 row-blocks x 16 column strips ---
  proj_gemm_kernel<<<(MM / 256) * (CC / 64), 256, 0, stream>>>(
      dy, wTp, b_proj, out);
}
